// Dynamic_conv1d_refer_51771535786361
// MI455X (gfx1250) — matrix operand order repaired and checked by reading, not compiled, not run on hardware
//
#include <hip/hip_runtime.h>

// ---------------------------------------------------------------------------
// Dynamic conv1d (attention-mixed kernel bank) for MI455X / gfx1250
//   B=32, IN=OUT=256, KS=3, L=4096, HIDDEN=65, K=4 banks, pad=1
//   Main conv done as per-sample GEMM: [256 x 768] x [768 x 4096]
//   using v_wmma_f32_16x16x32_f16 (f16 in, f32 accumulate).
//   Register blocking: 8 waves/block, each wave owns one 16-row M tile and
//   four 16-col N tiles -> every A fragment feeds 4 WMMAs.
// ---------------------------------------------------------------------------

typedef _Float16 v16h __attribute__((ext_vector_type(16)));
typedef _Float16 v8h  __attribute__((ext_vector_type(8)));
typedef float    v8f  __attribute__((ext_vector_type(8)));

#define NB      32
#define REFP    256
#define INP     256
#define OUTP    256
#define KSZ     3
#define NBANK   4
#define HIDDEN  65
#define LLEN    4096
#define LREF    64
#define TEMP    34.0f

#define BM      128         // output-channel tile per block (8 waves x 16)
#define BN      64          // length tile per block (4 x 16 per wave)
#define BK      32          // K chunk (channels per tap per WMMA step)
#define KTOT    (INP * KSZ) // 768
#define NCHUNK  (KTOT / BK) // 24

#define XROWS   (BN + 2)    // 66 (halo for ks=3, pad=1)
#define XPITCH  (INP + 8)   // 264 f16 -> 528B rows (16B aligned)

// ---------------------------------------------------------------------------
// Kernel A: attention MLP -> attn[B,4], agg_b[B,256]
// one block per batch sample, 256 threads
// ---------------------------------------------------------------------------
__global__ void attn_kernel(const float* __restrict__ ref,
                            const float* __restrict__ w1,
                            const float* __restrict__ w2,
                            const float* __restrict__ b2,
                            const float* __restrict__ bias,
                            float* __restrict__ attn_out,
                            float* __restrict__ aggb_out) {
    __shared__ float pooled[REFP];
    __shared__ float hbuf[HIDDEN];
    __shared__ float sm[NBANK];

    const int b   = blockIdx.x;
    const int tid = threadIdx.x;

    // global average pool over L_REF
    {
        const float* r = ref + ((size_t)(b * REFP + tid)) * LREF;
        float s = 0.f;
        #pragma unroll 8
        for (int i = 0; i < LREF; ++i) s += r[i];
        pooled[tid] = s * (1.0f / (float)LREF);
    }
    __syncthreads();

    // fc1 + relu
    if (tid < HIDDEN) {
        const float* wrow = w1 + (size_t)tid * REFP;
        float acc = 0.f;
        #pragma unroll 8
        for (int p = 0; p < REFP; ++p) acc += pooled[p] * wrow[p];
        hbuf[tid] = fmaxf(acc, 0.f);
    }
    __syncthreads();

    // fc2 + bias, /TEMP
    if (tid < NBANK) {
        const float* wrow = w2 + tid * HIDDEN;
        float acc = b2[tid];
        for (int j = 0; j < HIDDEN; ++j) acc += hbuf[j] * wrow[j];
        sm[tid] = acc * (1.0f / TEMP);
    }
    __syncthreads();

    if (tid == 0) {
        float m = sm[0];
        for (int k = 1; k < NBANK; ++k) m = fmaxf(m, sm[k]);
        float s = 0.f, e[NBANK];
        for (int k = 0; k < NBANK; ++k) { e[k] = __expf(sm[k] - m); s += e[k]; }
        float inv = 1.0f / s;
        for (int k = 0; k < NBANK; ++k) sm[k] = e[k] * inv;
    }
    __syncthreads();

    if (tid < NBANK) attn_out[b * NBANK + tid] = sm[tid];

    // agg_b = attn @ bias   (o = tid)
    {
        float ab = 0.f;
        for (int k = 0; k < NBANK; ++k) ab += sm[k] * bias[k * OUTP + tid];
        aggb_out[b * OUTP + tid] = ab;
    }
}

// ---------------------------------------------------------------------------
// Kernel B: aggregate kernel banks -> f16, GEMM-friendly layout
//   agg[b][o][s*256 + i] = (f16) sum_k attn[b][k] * weight[k][o][i][s]
// grid (OUTP, NB), 256 threads
// ---------------------------------------------------------------------------
__global__ void aggw_kernel(const float* __restrict__ weight,
                            const float* __restrict__ attn,
                            _Float16* __restrict__ agg) {
    __shared__ float a4[NBANK];
    const int o = blockIdx.x;
    const int b = blockIdx.y;
    const int tid = threadIdx.x;
    if (tid < NBANK) a4[tid] = attn[b * NBANK + tid];
    __syncthreads();

    _Float16* dst = agg + ((size_t)(b * OUTP + o)) * KTOT;
    const float* src0 = weight + (size_t)o * (INP * KSZ);
    #pragma unroll
    for (int e = tid; e < KTOT; e += 256) {
        const int s = e >> 8;          // tap
        const int i = e & 255;         // input channel
        const float* w = src0 + i * KSZ + s;
        float acc = a4[0] * w[0]
                  + a4[1] * w[1 * OUTP * INP * KSZ]
                  + a4[2] * w[2 * OUTP * INP * KSZ]
                  + a4[3] * w[3 * OUTP * INP * KSZ];
        dst[e] = (_Float16)acc;
    }
}

// ---------------------------------------------------------------------------
// Kernel C: main conv-as-GEMM with WMMA f16->f32
// grid (L/BN, OUTP/BM, NB), 256 threads (8 waves)
// ---------------------------------------------------------------------------
__global__ __launch_bounds__(256)
void dynconv_wmma_kernel(const float* __restrict__ x,
                         const _Float16* __restrict__ agg,
                         const float* __restrict__ aggb,
                         float* __restrict__ out) {
    __shared__ _Float16 xT[XROWS * XPITCH];   // [n][c] transposed x tile, f16

    const int n0 = blockIdx.x * BN;   // length offset
    const int m0 = blockIdx.y * BM;   // out-channel offset
    const int b  = blockIdx.z;

    const int tid  = threadIdx.x;
    const int wave = tid >> 5;
    const int lane = tid & 31;
    const int laneN  = lane & 15;
    const int laneHi = lane >> 4;     // 0 or 1

    // ---- stage x tile into LDS, transposed + f16 ----
    {
        const float* xb = x + (size_t)b * INP * LLEN;
        #pragma unroll
        for (int it = 0; it < (XROWS * INP) / 256; ++it) {
            const int e   = tid + it * 256;
            const int c   = e / XROWS;
            const int row = e - c * XROWS;
            const int l   = n0 - 1 + row;
            float v = 0.f;
            if (l >= 0 && l < LLEN) v = xb[(size_t)c * LLEN + l];
            xT[row * XPITCH + c] = (_Float16)v;
        }
    }
    __syncthreads();

    // wave -> one M tile; all four N tiles of the block
    const int m16 = wave * 16;

    // per-lane A row base (agg layout: [b][o][s*256+i], contiguous K)
    const int mrow = m0 + m16 + laneN;
    const _Float16* aRow = agg + ((size_t)(b * OUTP + mrow)) * KTOT;
    const int aKoff = laneHi * 16;    // lanes<16: K 0..15, else 16..31
    const int bKoff = laneHi * 16;    // lanes<16: K 0..15, else 16..31
    const int bLane = laneN * XPITCH + bKoff;

    v8f c0 = {}, c1 = {}, c2 = {}, c3 = {};

    #pragma unroll 8
    for (int kc = 0; kc < NCHUNK; ++kc) {
        const int s  = kc >> 3;              // tap 0..2
        const int i0 = (kc & 7) << 5;        // channel chunk base
        const int kbase = s * INP + i0;

        // A fragment: 2 x 16B contiguous global loads, reused by 4 WMMAs
        union { v16h v; v8h h[2]; } A;
        A.h[0] = *(const v8h*)(aRow + kbase + aKoff);
        A.h[1] = *(const v8h*)(aRow + kbase + 8 + aKoff);

        // B fragments from LDS (transposed tile): 2 x b128 each
        const int rb = bLane + s * XPITCH + i0;
        union { v16h v; v8h h[2]; } B0, B1, B2, B3;
        B0.h[0] = *(const v8h*)(&xT[rb]);
        B0.h[1] = *(const v8h*)(&xT[rb + 8]);
        B1.h[0] = *(const v8h*)(&xT[rb + 16 * XPITCH]);
        B1.h[1] = *(const v8h*)(&xT[rb + 16 * XPITCH + 8]);
        B2.h[0] = *(const v8h*)(&xT[rb + 32 * XPITCH]);
        B2.h[1] = *(const v8h*)(&xT[rb + 32 * XPITCH + 8]);
        B3.h[0] = *(const v8h*)(&xT[rb + 48 * XPITCH]);
        B3.h[1] = *(const v8h*)(&xT[rb + 48 * XPITCH + 8]);

        c0 = __builtin_amdgcn_wmma_f32_16x16x32_f16(false, A.v, false, B0.v,
                                                    (short)0, c0, false, false);
        c1 = __builtin_amdgcn_wmma_f32_16x16x32_f16(false, A.v, false, B1.v,
                                                    (short)0, c1, false, false);
        c2 = __builtin_amdgcn_wmma_f32_16x16x32_f16(false, A.v, false, B2.v,
                                                    (short)0, c2, false, false);
        c3 = __builtin_amdgcn_wmma_f32_16x16x32_f16(false, A.v, false, B3.v,
                                                    (short)0, c3, false, false);
    }

    // ---- epilogue: C layout = lanes 0-15 N=lane M=r, lanes 16-31 N=lane-16 M=r+8
    const int col0 = n0 + laneN;
    const int rbase = laneHi * 8;
    #pragma unroll
    for (int r = 0; r < 8; ++r) {
        const int M = m0 + m16 + r + rbase;
        const float bb = aggb[b * OUTP + M];
        float* orow = out + ((size_t)(b * OUTP + M)) * LLEN;
        orow[col0]      = c0[r] + bb;
        orow[col0 + 16] = c1[r] + bb;
        orow[col0 + 32] = c2[r] + bb;
        orow[col0 + 48] = c3[r] + bb;
    }
}

// ---------------------------------------------------------------------------
// launch
// ---------------------------------------------------------------------------
extern "C" void kernel_launch(void* const* d_in, const int* in_sizes, int n_in,
                              void* d_out, int out_size, void* d_ws, size_t ws_size,
                              hipStream_t stream) {
    const float* ref    = (const float*)d_in[0];
    const float* x      = (const float*)d_in[1];
    const float* att_w1 = (const float*)d_in[2];
    const float* att_w2 = (const float*)d_in[3];
    const float* att_b2 = (const float*)d_in[4];
    const float* weight = (const float*)d_in[5];
    const float* bias   = (const float*)d_in[6];
    float* out = (float*)d_out;

    // workspace layout
    const size_t aggw_elems = (size_t)NB * OUTP * KTOT;       // 6.29M f16
    char* ws = (char*)d_ws;
    _Float16* agg  = (_Float16*)ws;
    float*    attn = (float*)(ws + aggw_elems * sizeof(_Float16));
    float*    aggb = attn + NB * NBANK;

    attn_kernel<<<NB, 256, 0, stream>>>(ref, att_w1, att_w2, att_b2, bias,
                                        attn, aggb);
    aggw_kernel<<<dim3(OUTP, NB), 256, 0, stream>>>(weight, attn, agg);
    dynconv_wmma_kernel<<<dim3(LLEN / BN, OUTP / BM, NB), 256, 0, stream>>>(
        x, agg, aggb, out);
}
